// ChannelAttention_9740985827411
// MI455X (gfx1250) — matrix operand order repaired and checked by reading, not compiled, not run on hardware
//
#include <hip/hip_runtime.h>

// ---------------- types ----------------
typedef __bf16 bf16;
typedef __attribute__((ext_vector_type(16))) __bf16    v16bf;
typedef __attribute__((ext_vector_type(8)))  float     v8f;
typedef __attribute__((ext_vector_type(4)))  unsigned  u32x4;

union FragAB { v16bf v; u32x4 u[2]; };   // 32 bytes: one WMMA A or B operand
union Chunk  { u32x4 u; bf16 h[8]; };    // 16 bytes: 8 bf16

// ---------------- problem constants ----------------
#define Bb   8
#define Nn   4096
#define Cc   768
#define Gg   8
#define GD   96           // Cc / Gg
#define M1   (Bb * Nn)    // 32768 rows of x
#define N1   (3 * Cc)     // 2304 qkv cols
#define LDS_PAD 40        // bf16 per LDS row (80B: 16B-aligned, conflict-free)

// ---------------- conversion kernels ----------------
__global__ void cvt_f32_bf16(const float* __restrict__ in, bf16* __restrict__ out, int n) {
  int i = blockIdx.x * blockDim.x + threadIdx.x;
  int stride = gridDim.x * blockDim.x;
  for (; i < n; i += stride) out[i] = (bf16)in[i];
}

// out[c*rows + r] = (bf16) in[r*cols + c]   (produce N-major, K-contiguous weights)
__global__ void transpose_f32_bf16(const float* __restrict__ in, bf16* __restrict__ out,
                                   int rows, int cols) {
  long i = (long)blockIdx.x * blockDim.x + threadIdx.x;
  long n = (long)rows * cols;
  long stride = (long)gridDim.x * blockDim.x;
  for (; i < n; i += stride) {
    long r = i / cols, c = i % cols;
    out[c * (long)rows + r] = (bf16)in[i];
  }
}

// ---------------- generic 128x128 WMMA bf16 GEMM ----------------
// C[m][n] = sum_k A[m][k] * Bt[n][k] (+ bias[n]); A: MxK row-major, Bt: NxK row-major.
// grid = (M/128, N/128), block = 256 (8 waves; wave tile 32x64 = 2x4 WMMA tiles).
template <bool OUT_BF16>
__global__ __launch_bounds__(256)
void gemm_bf16_128x128(const bf16* __restrict__ A, int lda,
                       const bf16* __restrict__ Bt, int ldb,
                       const float* __restrict__ bias,
                       void* __restrict__ Out, int ldo, int K) {
  __shared__ bf16 As[128][LDS_PAD];
  __shared__ bf16 Bs[128][LDS_PAD];
  const int t    = threadIdx.x;
  const int lane = t & 31;
  const int wid  = t >> 5;
  const int m0   = blockIdx.x * 128;
  const int n0   = blockIdx.y * 128;
  const int wm   = (wid & 3) * 32;   // wave M offset within block tile
  const int wn   = (wid >> 2) * 64;  // wave N offset

  v8f acc[2][4] = {};

  for (int k0 = 0; k0 < K; k0 += 32) {
#pragma unroll
    for (int i = 0; i < 2; i++) {                 // 512 16B chunks / 256 threads
      int id = t + i * 256;
      int r  = id >> 2;
      int cc = (id & 3) << 3;
      const bf16* ga = A  + (long)(m0 + r) * lda + k0 + cc;
      const bf16* gb = Bt + (long)(n0 + r) * ldb + k0 + cc;
      *(u32x4*)&As[r][cc] = *(const u32x4*)ga;
      *(u32x4*)&Bs[r][cc] = *(const u32x4*)gb;
      if (k0 + 32 < K) {                          // hint next K tile into cache
        __builtin_prefetch(ga + 32, 0, 1);
        __builtin_prefetch(gb + 32, 0, 1);
      }
    }
    __syncthreads();

    const int fr = lane & 15;
    const int ka = (lane >> 4) * 8;    // A: K in {ka..ka+7} U {ka+16..ka+23}
    const int kb = (lane >> 4) * 8;    // B: K in {kb..kb+7} U {kb+16..kb+23}
    FragAB fa[2], fb[4];
#pragma unroll
    for (int i = 0; i < 2; i++) {
      const bf16* p = &As[wm + i * 16 + fr][0];
      fa[i].u[0] = *(const u32x4*)(p + ka);
      fa[i].u[1] = *(const u32x4*)(p + ka + 16);
    }
#pragma unroll
    for (int j = 0; j < 4; j++) {
      const bf16* p = &Bs[wn + j * 16 + fr][0];
      fb[j].u[0] = *(const u32x4*)(p + kb);
      fb[j].u[1] = *(const u32x4*)(p + kb + 16);
    }
#pragma unroll
    for (int i = 0; i < 2; i++)
#pragma unroll
      for (int j = 0; j < 4; j++)
        acc[i][j] = __builtin_amdgcn_wmma_f32_16x16x32_bf16(
            false, fa[i].v, false, fb[j].v, (short)0, acc[i][j], false, false);
    __syncthreads();
  }

  const int col   = lane & 15;
  const int rbase = (lane >> 4) * 8;   // C/D layout: VGPR e -> row e (+8 for hi lanes)
#pragma unroll
  for (int i = 0; i < 2; i++)
#pragma unroll
    for (int j = 0; j < 4; j++) {
      int gn = n0 + wn + j * 16 + col;
      float bv = bias ? bias[gn] : 0.0f;
#pragma unroll
      for (int e = 0; e < 8; e++) {
        int gm = m0 + wm + i * 16 + rbase + e;
        float val = acc[i][j][e] + bv;
        if constexpr (OUT_BF16) ((bf16*)Out)[(long)gm * ldo + gn] = (bf16)val;
        else                    ((float*)Out)[(long)gm * ldo + gn] = val;
      }
    }
}

// ---------------- attention logits: S = scale * q^T k  (per b,g) ----------------
// grid = 64 (b*g), block = 128 (4 waves; wave tile 48x48 = 3x3 WMMA tiles), K = 4096 tokens.
__global__ __launch_bounds__(128)
void attn_qtk(const bf16* __restrict__ qkv, float* __restrict__ S, float scale) {
  __shared__ bf16 Qs[GD][LDS_PAD];   // [channel][token_k]
  __shared__ bf16 Ks[GD][LDS_PAD];
  const int t = threadIdx.x, lane = t & 31, wid = t >> 5;
  const int bg = blockIdx.x, b = bg >> 3, g = bg & 7;
  const long base = (long)b * Nn * N1;
  const int  qoff = g * GD;
  const int  koff = Cc + g * GD;
  const int  wm = (wid & 1) * 48;
  const int  wn = (wid >> 1) * 48;

  v8f acc[3][3] = {};

  for (int k0 = 0; k0 < Nn; k0 += 32) {
#pragma unroll
    for (int i = 0; i < 3; i++) {           // 32 tok x 12 chunks = 384 / 128 thr
      int id = t + i * 128;
      int nl = id / 12;
      int cj = (id % 12) * 8;
      const long rowb = base + (long)(k0 + nl) * N1;
      Chunk cq, ck;
      cq.u = *(const u32x4*)(qkv + rowb + qoff + cj);
      ck.u = *(const u32x4*)(qkv + rowb + koff + cj);
#pragma unroll
      for (int e = 0; e < 8; e++) {         // transpose into channel-major LDS
        Qs[cj + e][nl] = cq.h[e];
        Ks[cj + e][nl] = ck.h[e];
      }
    }
    __syncthreads();

    const int fr = lane & 15;
    const int ka = (lane >> 4) * 8;
    const int kb = (lane >> 4) * 8;
    FragAB fa[3], fb[3];
#pragma unroll
    for (int i = 0; i < 3; i++) {
      const bf16* pq = &Qs[wm + i * 16 + fr][0];
      fa[i].u[0] = *(const u32x4*)(pq + ka);
      fa[i].u[1] = *(const u32x4*)(pq + ka + 16);
      const bf16* pk = &Ks[wn + i * 16 + fr][0];
      fb[i].u[0] = *(const u32x4*)(pk + kb);
      fb[i].u[1] = *(const u32x4*)(pk + kb + 16);
    }
#pragma unroll
    for (int i = 0; i < 3; i++)
#pragma unroll
      for (int j = 0; j < 3; j++)
        acc[i][j] = __builtin_amdgcn_wmma_f32_16x16x32_bf16(
            false, fa[i].v, false, fb[j].v, (short)0, acc[i][j], false, false);
    __syncthreads();
  }

  float* Sp = S + (long)bg * GD * GD;
  const int col = lane & 15, rbase = (lane >> 4) * 8;
#pragma unroll
  for (int i = 0; i < 3; i++)
#pragma unroll
    for (int j = 0; j < 3; j++)
#pragma unroll
      for (int e = 0; e < 8; e++)
        Sp[(wm + i * 16 + rbase + e) * GD + wn + j * 16 + col] = acc[i][j][e] * scale;
}

// ---------------- softmax over rows of 96 (one wave32 per row) ----------------
__global__ __launch_bounds__(256)
void softmax_rows(const float* __restrict__ S, bf16* __restrict__ A) {
  const int lane = threadIdx.x & 31;
  const int wid  = threadIdx.x >> 5;
  const int row  = blockIdx.x * 8 + wid;          // 64*96 = 6144 rows total
  const float* p = S + (long)row * GD;
  float v0 = p[lane], v1 = p[lane + 32], v2 = p[lane + 64];
  float m = fmaxf(v0, fmaxf(v1, v2));
#pragma unroll
  for (int off = 16; off > 0; off >>= 1) m = fmaxf(m, __shfl_xor(m, off, 32));
  float e0 = __expf(v0 - m), e1 = __expf(v1 - m), e2 = __expf(v2 - m);
  float s = e0 + e1 + e2;
#pragma unroll
  for (int off = 16; off > 0; off >>= 1) s += __shfl_xor(s, off, 32);
  float inv = 1.0f / s;
  bf16* o = A + (long)row * GD;
  o[lane]      = (bf16)(e0 * inv);
  o[lane + 32] = (bf16)(e1 * inv);
  o[lane + 64] = (bf16)(e2 * inv);
}

// ---------------- out1 = v @ attn^T  (per b,g; tokens x 96) ----------------
// grid = (32, 64), block = 256 (8 waves; wave tile 32x48 = 2x3 WMMA tiles), K = 96.
__global__ __launch_bounds__(256)
void attn_av(const bf16* __restrict__ qkv, const bf16* __restrict__ At,
             bf16* __restrict__ Out) {
  __shared__ bf16 Vs[128][LDS_PAD];
  __shared__ bf16 Bs[GD][LDS_PAD];
  const int t = threadIdx.x, lane = t & 31, wid = t >> 5;
  const int m0 = blockIdx.x * 128;
  const int bg = blockIdx.y, b = bg >> 3, g = bg & 7;
  const long vbase = (long)b * Nn * N1 + 2 * Cc + g * GD;
  const bf16* attn = At + (long)bg * GD * GD;
  const int wm = (wid & 3) * 32;
  const int wn = (wid >> 2) * 48;

  v8f acc[2][3] = {};

  for (int k0 = 0; k0 < GD; k0 += 32) {
#pragma unroll
    for (int i = 0; i < 2; i++) {               // V tile: 128 tok x 32 d
      int id = t + i * 256;
      int r = id >> 2, cc = (id & 3) << 3;
      *(u32x4*)&Vs[r][cc] = *(const u32x4*)(qkv + vbase + (long)(m0 + r) * N1 + k0 + cc);
    }
    for (int id = t; id < GD * 4; id += 256) {  // attn tile: 96 x 32
      int r = id >> 2, cc = (id & 3) << 3;
      *(u32x4*)&Bs[r][cc] = *(const u32x4*)(attn + (long)r * GD + k0 + cc);
    }
    __syncthreads();

    const int fr = lane & 15;
    const int ka = (lane >> 4) * 8;
    const int kb = (lane >> 4) * 8;
    FragAB fa[2], fb[3];
#pragma unroll
    for (int i = 0; i < 2; i++) {
      const bf16* p = &Vs[wm + i * 16 + fr][0];
      fa[i].u[0] = *(const u32x4*)(p + ka);
      fa[i].u[1] = *(const u32x4*)(p + ka + 16);
    }
#pragma unroll
    for (int j = 0; j < 3; j++) {
      const bf16* p = &Bs[wn + j * 16 + fr][0];
      fb[j].u[0] = *(const u32x4*)(p + kb);
      fb[j].u[1] = *(const u32x4*)(p + kb + 16);
    }
#pragma unroll
    for (int i = 0; i < 2; i++)
#pragma unroll
      for (int j = 0; j < 3; j++)
        acc[i][j] = __builtin_amdgcn_wmma_f32_16x16x32_bf16(
            false, fa[i].v, false, fb[j].v, (short)0, acc[i][j], false, false);
    __syncthreads();
  }

  const int col = lane & 15, rbase = (lane >> 4) * 8;
  const long orow0 = (long)b * Nn + m0;
#pragma unroll
  for (int i = 0; i < 2; i++)
#pragma unroll
    for (int j = 0; j < 3; j++)
#pragma unroll
      for (int e = 0; e < 8; e++)
        Out[(orow0 + wm + i * 16 + rbase + e) * Cc + g * GD + wn + j * 16 + col] =
            (bf16)acc[i][j][e];
}

// ---------------- host launcher ----------------
extern "C" void kernel_launch(void* const* d_in, const int* in_sizes, int n_in,
                              void* d_out, int out_size, void* d_ws, size_t ws_size,
                              hipStream_t stream) {
  (void)in_sizes; (void)n_in; (void)out_size; (void)ws_size;
  const float* x      = (const float*)d_in[0];
  const float* qkv_w  = (const float*)d_in[1];
  const float* qkv_b  = (const float*)d_in[2];
  const float* proj_w = (const float*)d_in[3];
  const float* proj_b = (const float*)d_in[4];
  // d_in[5] = size (unused by reference math)

  char* ws = (char*)d_ws;
  bf16*  xb   = (bf16*)(ws);                         // 32768x768   bf16 (50.3 MB)
  bf16*  wt   = (bf16*)(ws + 50331648);              // 2304x768    bf16 ( 3.5 MB)
  bf16*  pt   = (bf16*)(ws + 53870592);              // 768x768     bf16 ( 1.2 MB)
  bf16*  qkv  = (bf16*)(ws + 55050240);              // 32768x2304  bf16 (151 MB)
  float* S    = (float*)(ws + 206045184);            // 64x96x96    f32  ( 2.4 MB)
  bf16*  attn = (bf16*)(ws + 208404480);             // 64x96x96    bf16 ( 1.2 MB)
  bf16*  out1 = (bf16*)(ws + 209584128);             // 32768x768   bf16 (50.3 MB)

  cvt_f32_bf16<<<1024, 256, 0, stream>>>(x, xb, M1 * Cc);
  transpose_f32_bf16<<<1024, 256, 0, stream>>>(qkv_w, wt, Cc, N1);
  transpose_f32_bf16<<<512, 256, 0, stream>>>(proj_w, pt, Cc, Cc);

  // qkv = x @ qkv_w + qkv_b   (bf16 out, ld 2304)
  gemm_bf16_128x128<true><<<dim3(M1 / 128, N1 / 128), 256, 0, stream>>>(
      xb, Cc, wt, Cc, qkv_b, qkv, N1, Cc);

  // S = (1/sqrt(N)) * q^T k per (b,g)
  attn_qtk<<<64, 128, 0, stream>>>(qkv, S, 0.015625f);

  // softmax over last axis
  softmax_rows<<<768, 256, 0, stream>>>(S, attn);

  // out1[b, n, g*96+c] = sum_d attn[c,d] * v[n,d]
  attn_av<<<dim3(32, 64), 256, 0, stream>>>(qkv, attn, out1);

  // d_out = out1 @ proj_w + proj_b  (f32 out)
  gemm_bf16_128x128<false><<<dim3(M1 / 128, Cc / 128), 256, 0, stream>>>(
      out1, Cc, pt, Cc, proj_b, d_out, Cc, Cc);
}
